// TrajectoryPredictor_56040733278781
// MI455X (gfx1250) — matrix operand order repaired and checked by reading, not compiled, not run on hardware
//
#include <hip/hip_runtime.h>

// ---------------- problem constants (match reference) ----------------
constexpr int B  = 8192;
constexpr int S  = 100;
constexpr int I  = 15;
constexpr int H  = 128;
constexpr int O  = 2;
constexpr int G4 = 4 * H;            // 512 gate columns
constexpr int BT = 16;               // batch tile per workgroup (WMMA M)
constexpr int NT = G4 / 16;          // 32 N-tiles of 16
constexpr int KT = H / 32;           // 4 K-steps of 32 (bf16 WMMA K)
constexpr int THREADS = 256;         // 8 wave32 -> 2 waves/SIMD -> 512-VGPR budget
constexpr int WAVES = THREADS / 32;
constexpr int NTW = NT / WAVES;      // 4 N-tiles per wave

// fragment-ordered weight buffer sizes (elements)
constexpr int FRAG_IH0 = NT * 1  * 32 * 16;  // 16384  (K padded 15->32)
constexpr int FRAG_HH  = NT * KT * 32 * 16;  // 65536  (K=128)

// workspace byte offsets
constexpr size_t OFF_IH0 = 0;
constexpr size_t OFF_HH0 = OFF_IH0 + (size_t)FRAG_IH0 * 2;
constexpr size_t OFF_IH1 = OFF_HH0 + (size_t)FRAG_HH * 2;
constexpr size_t OFF_HH1 = OFF_IH1 + (size_t)FRAG_HH * 2;
constexpr size_t OFF_B0  = OFF_HH1 + (size_t)FRAG_HH * 2;
constexpr size_t OFF_B1  = OFF_B0  + (size_t)G4 * 4;

typedef __attribute__((ext_vector_type(16))) __bf16 v16bf;
typedef __attribute__((ext_vector_type(8)))  float  v8f;

#define WMMA_BF16(A, Bm, C) \
  __builtin_amdgcn_wmma_f32_16x16x32_bf16(false, (A), false, (Bm), (short)0, (C), false, false)

// fast activations on the hardware transcendental pipe
__device__ __forceinline__ float fast_sigmoid(float x) {
  float e = __builtin_amdgcn_exp2f(-1.44269504f * x);
  return __builtin_amdgcn_rcpf(1.0f + e);
}
__device__ __forceinline__ float fast_tanh(float x) {
  return 2.0f * fast_sigmoid(2.0f * x) - 1.0f;
}

// ---- A-fragment address for 16-bit A 16x32 tiles (ISA 7.12.2) ----
__device__ __forceinline__ int afrag_off(int m, int k) {
  int kt   = k >> 5;
  int ko   = k & 31;
  int lane = m + ((ko & 8) ? 16 : 0);
  int idx  = (ko & 7) | ((ko & 16) >> 1);
  return (kt * 32 + lane) * 16 + idx;
}

// ================= setup: shuffle fp32 weights -> bf16 B-fragments =================
// B element (k,n) within a 32x16 tile: lane = (n&15) + 16*((k>>3)&1), idx = (k&7) + 8*(k>=16)
__device__ __forceinline__ void frag_hh(const float* __restrict__ w, __bf16* __restrict__ f, int idx) {
  int e    = idx & 15;
  int lane = (idx >> 4) & 31;
  int kt   = (idx >> 9) & (KT - 1);
  int nt   = idx >> 11;
  int n    = nt * 16 + (lane & 15);
  int k    = kt * 32 + ((lane >> 4) << 3) + (e & 7) + ((e >> 3) << 4);
  f[idx]   = (__bf16)w[n * H + k];
}

__global__ __launch_bounds__(256)
void setup_kernel(const float* __restrict__ w_ih0, const float* __restrict__ w_hh0,
                  const float* __restrict__ b_ih0, const float* __restrict__ b_hh0,
                  const float* __restrict__ w_ih1, const float* __restrict__ w_hh1,
                  const float* __restrict__ b_ih1, const float* __restrict__ b_hh1,
                  __bf16* fIh0, __bf16* fHh0, __bf16* fIh1, __bf16* fHh1,
                  float* b0, float* b1) {
  int idx = blockIdx.x * 256 + threadIdx.x;
  if (idx < FRAG_IH0) {                      // w_ih0 [512 x 15] zero-padded to K=32
    int e    = idx & 15;
    int lane = (idx >> 4) & 31;
    int nt   = idx >> 9;
    int n    = nt * 16 + (lane & 15);
    int k    = ((lane >> 4) << 3) + (e & 7) + ((e >> 3) << 4);
    fIh0[idx] = (__bf16)((k < I) ? w_ih0[n * I + k] : 0.0f);
    return;
  }
  idx -= FRAG_IH0;
  if (idx < FRAG_HH) { frag_hh(w_hh0, fHh0, idx); return; }
  idx -= FRAG_HH;
  if (idx < FRAG_HH) { frag_hh(w_ih1, fIh1, idx); return; }
  idx -= FRAG_HH;
  if (idx < FRAG_HH) { frag_hh(w_hh1, fHh1, idx); return; }
  idx -= FRAG_HH;
  if (idx < G4) { b0[idx] = b_ih0[idx] + b_hh0[idx]; return; }
  idx -= G4;
  if (idx < G4) { b1[idx] = b_ih1[idx] + b_hh1[idx]; return; }
}

// ========== elementwise LSTM cell update; h stored back in A-fragment bf16 ==========
__device__ __forceinline__ void cell_update(const float* gates, float* cs,
                                            __bf16* hFrag, int tid) {
#pragma unroll
  for (int e = 0; e < (BT * H) / THREADS; ++e) {   // 8 elems/thread @ 256 threads
    int idx = tid + e * THREADS;
    int m = idx >> 7;
    int j = idx & (H - 1);
    float gi = gates[m * G4 + j];
    float gf = gates[m * G4 + H + j];
    float gg = gates[m * G4 + 2 * H + j];
    float go = gates[m * G4 + 3 * H + j];
    float c  = fast_sigmoid(gf) * cs[idx] + fast_sigmoid(gi) * fast_tanh(gg);
    cs[idx]  = c;
    hFrag[afrag_off(m, j)] = (__bf16)(fast_sigmoid(go) * fast_tanh(c));
  }
}

// ================= main: one workgroup owns a 16-row batch tile =================
// Weight-stationary for 44 of 52 B-fragments (352 VGPRs). The small w_ih0 and the
// last K-chunk of w_hh0 (both L2-resident) are streamed per step via global_load
// with an LICM-defeating barrier on 32-bit offsets, keeping worst-phase register
// demand ~478 < the 512/wave budget (2 waves/SIMD on the 1024-reg file).
__global__ __launch_bounds__(THREADS)
void lstm_traj_kernel(const float* __restrict__ x,
                      const __bf16* __restrict__ fIh0, const __bf16* __restrict__ fHh0,
                      const __bf16* __restrict__ fIh1, const __bf16* __restrict__ fHh1,
                      const float* __restrict__ b0, const float* __restrict__ b1,
                      const float* __restrict__ fc_w, const float* __restrict__ fc_b,
                      const int* __restrict__ fptr, float* __restrict__ out) {
  __shared__ __align__(32) __bf16 xA[32 * 16];           // 1 KB  (K=32 padded x tile)
  __shared__ __align__(32) __bf16 h0f[KT * 32 * 16];     // 4 KB  (h0 A-fragments)
  __shared__ __align__(32) __bf16 h1f[KT * 32 * 16];     // 4 KB  (h1 A-fragments)
  __shared__ float gates[BT * G4];                        // 32 KB
  __shared__ float c0s[BT * H];                           // 8 KB
  __shared__ float c1s[BT * H];                           // 8 KB
  __shared__ float inp[BT * 16];                          // 1 KB (decoder feedback)

  const int tid   = threadIdx.x;
  const int lane  = tid & 31;
  const int wv    = tid >> 5;
  const int bBase = blockIdx.x * BT;
  const int F     = fptr[0];

  // ---- load this wave's persistent weight fragments into registers ----
  const v16bf* gIh0 = reinterpret_cast<const v16bf*>(fIh0);
  const v16bf* gHh0 = reinterpret_cast<const v16bf*>(fHh0);
  const v16bf* gIh1 = reinterpret_cast<const v16bf*>(fIh1);
  const v16bf* gHh1 = reinterpret_cast<const v16bf*>(fHh1);

  v16bf wHh0[NTW][KT - 1], wIh1[NTW][KT], wHh1[NTW][KT];
  float bv0[NTW], bv1[NTW];
#pragma unroll
  for (int i = 0; i < NTW; ++i) {
    const int nt = wv * NTW + i;
#pragma unroll
    for (int kt = 0; kt < KT - 1; ++kt)
      wHh0[i][kt] = gHh0[(nt * KT + kt) * 32 + lane];
#pragma unroll
    for (int kt = 0; kt < KT; ++kt) {
      wIh1[i][kt] = gIh1[(nt * KT + kt) * 32 + lane];
      wHh1[i][kt] = gHh1[(nt * KT + kt) * 32 + lane];
    }
    bv0[i] = b0[nt * 16 + (lane & 15)];
    bv1[i] = b1[nt * 16 + (lane & 15)];
  }

  // zero state (xA padding lanes stay zero for the whole run: only k<15 rewritten)
  for (int i = tid; i < BT * H; i += THREADS) { c0s[i] = 0.0f; c1s[i] = 0.0f; }
  for (int i = tid; i < KT * 32 * 16; i += THREADS) { h0f[i] = (__bf16)0.0f; h1f[i] = (__bf16)0.0f; }
  for (int i = tid; i < 32 * 16; i += THREADS) xA[i] = (__bf16)0.0f;
  __syncthreads();

  const v16bf* xAv = reinterpret_cast<const v16bf*>(xA);
  const v16bf* h0v = reinterpret_cast<const v16bf*>(h0f);
  const v16bf* h1v = reinterpret_cast<const v16bf*>(h1f);
  const int colB  = lane & 15;
  const int mBase = (lane < 16) ? 0 : 8;

  // add per-lane scalar bias, then scatter C fragment to LDS gate buffer
  auto store_tile = [&](v8f acc, int nt, float bv) {
    const int col = nt * 16 + colB;
#pragma unroll
    for (int r = 0; r < 8; ++r)
      gates[(mBase + r) * G4 + col] = acc[r] + bv;
  };

  auto lstm_step = [&]() {
    // ---- layer 0: gates = h0*Whh0^T + xA*Wih0^T + b0 ----
    {
      // preload shared A-fragments once for both N-tile groups
      v16bf ah[KT];
#pragma unroll
      for (int kt = 0; kt < KT; ++kt) ah[kt] = h0v[kt * 32 + lane];
      v16bf ax = xAv[lane];
#pragma unroll
      for (int g = 0; g < NTW / 2; ++g) {
        const int i0 = 2 * g, i1 = 2 * g + 1;
        // stream w_ih0 + w_hh0[kt=3] (offset barriers defeat LICM; global_load)
        int o0 = (wv * NTW + i0) * 32 + lane;
        int o1 = o0 + 32;
        int q0 = ((wv * NTW + i0) * KT + (KT - 1)) * 32 + lane;
        int q1 = q0 + KT * 32;
        asm volatile("" : "+v"(o0), "+v"(o1), "+v"(q0), "+v"(q1));
        v16bf wx0 = gIh0[o0];
        v16bf wx1 = gIh0[o1];
        v16bf wq0 = gHh0[q0];
        v16bf wq1 = gHh0[q1];
        v8f a0 = {};
        v8f a1 = {};
#pragma unroll
        for (int kt = 0; kt < KT - 1; ++kt) {
          a0 = WMMA_BF16(ah[kt], wHh0[i0][kt], a0);
          a1 = WMMA_BF16(ah[kt], wHh0[i1][kt], a1);
        }
        a0 = WMMA_BF16(ah[KT - 1], wq0, a0);
        a1 = WMMA_BF16(ah[KT - 1], wq1, a1);
        a0 = WMMA_BF16(ax, wx0, a0);
        a1 = WMMA_BF16(ax, wx1, a1);
        store_tile(a0, wv * NTW + i0, bv0[i0]);
        store_tile(a1, wv * NTW + i1, bv0[i1]);
      }
    }
    __syncthreads();
    cell_update(gates, c0s, h0f, tid);
    __syncthreads();
    // ---- layer 1: gates = h0*Wih1^T + h1*Whh1^T + b1 ----
    {
      v16bf ah0[KT];
#pragma unroll
      for (int kt = 0; kt < KT; ++kt) ah0[kt] = h0v[kt * 32 + lane];
#pragma unroll
      for (int g = 0; g < NTW / 2; ++g) {
        const int i0 = 2 * g, i1 = 2 * g + 1;
        v8f a0 = {};
        v8f a1 = {};
#pragma unroll
        for (int kt = 0; kt < KT; ++kt) {
          a0 = WMMA_BF16(ah0[kt], wIh1[i0][kt], a0);
          a1 = WMMA_BF16(ah0[kt], wIh1[i1][kt], a1);
        }
#pragma unroll
        for (int kt = 0; kt < KT; ++kt) {      // h1 frags loaded in-loop (VGPR headroom)
          v16bf a = h1v[kt * 32 + lane];
          a0 = WMMA_BF16(a, wHh1[i0][kt], a0);
          a1 = WMMA_BF16(a, wHh1[i1][kt], a1);
        }
        store_tile(a0, wv * NTW + i0, bv1[i0]);
        store_tile(a1, wv * NTW + i1, bv1[i1]);
      }
    }
    __syncthreads();
    cell_update(gates, c1s, h1f, tid);
    __syncthreads();
  };

  // ---------------- encoder over S observed frames ----------------
  for (int t = 0; t < S; ++t) {
    for (int i = tid; i < BT * I; i += THREADS) {
      int m = i / I, k = i - m * I;
      xA[afrag_off(m, k)] = (__bf16)x[((bBase + m) * S + t) * I + k];
    }
    __syncthreads();
    lstm_step();
  }

  // ---------------- autoregressive decoder ----------------
  for (int i = tid; i < BT * I; i += THREADS) {
    int m = i / I, k = i - m * I;
    inp[m * 16 + k] = x[((bBase + m) * S + (S - 1)) * I + k];
  }
  __syncthreads();

  for (int t = 0; t < F; ++t) {
    for (int i = tid; i < BT * I; i += THREADS) {
      int m = i / I, k = i - m * I;
      xA[afrag_off(m, k)] = (__bf16)inp[m * 16 + k];
    }
    __syncthreads();
    lstm_step();
    // tiny FC head: pred[16x2] = h1 @ fc_w.T + fc_b ; write output + feedback
    if (tid < BT * O) {
      int row = tid >> 1, oc = tid & 1;
      float sum = fc_b[oc];
      for (int k = 0; k < H; ++k)
        sum += fc_w[oc * H + k] * (float)h1f[afrag_off(row, k)];
      out[((bBase + row) * F + t) * O + oc] = sum;
      inp[row * 16 + oc] = sum;
    }
    __syncthreads();
  }
}

// ================= host launcher =================
extern "C" void kernel_launch(void* const* d_in, const int* in_sizes, int n_in,
                              void* d_out, int out_size, void* d_ws, size_t ws_size,
                              hipStream_t stream) {
  const float* x     = (const float*)d_in[0];
  const float* w_ih0 = (const float*)d_in[1];
  const float* w_hh0 = (const float*)d_in[2];
  const float* b_ih0 = (const float*)d_in[3];
  const float* b_hh0 = (const float*)d_in[4];
  const float* w_ih1 = (const float*)d_in[5];
  const float* w_hh1 = (const float*)d_in[6];
  const float* b_ih1 = (const float*)d_in[7];
  const float* b_hh1 = (const float*)d_in[8];
  const float* fc_w  = (const float*)d_in[9];
  const float* fc_b  = (const float*)d_in[10];
  const int*   fptr  = (const int*)d_in[11];

  char* ws = (char*)d_ws;
  __bf16* fIh0 = (__bf16*)(ws + OFF_IH0);
  __bf16* fHh0 = (__bf16*)(ws + OFF_HH0);
  __bf16* fIh1 = (__bf16*)(ws + OFF_IH1);
  __bf16* fHh1 = (__bf16*)(ws + OFF_HH1);
  float*  b0   = (float*)(ws + OFF_B0);
  float*  b1   = (float*)(ws + OFF_B1);

  const int setupTotal = FRAG_IH0 + 3 * FRAG_HH + 2 * G4;
  setup_kernel<<<(setupTotal + 255) / 256, 256, 0, stream>>>(
      w_ih0, w_hh0, b_ih0, b_hh0, w_ih1, w_hh1, b_ih1, b_hh1,
      fIh0, fHh0, fIh1, fHh1, b0, b1);

  lstm_traj_kernel<<<B / BT, THREADS, 0, stream>>>(
      x, fIh0, fHh0, fIh1, fHh1, b0, b1, fc_w, fc_b, fptr, (float*)d_out);
}
